// GATClassifier_45543833207124
// MI455X (gfx1250) — matrix operand order repaired and checked by reading, not compiled, not run on hardware
//
#include <hip/hip_runtime.h>

#define NN   10000
#define EE   160000
#define ENL  (EE + NN)
#define INC  128
#define HH   8
#define DD   64
#define HD   512
#define CC   2

typedef __bf16 bf16_t;
typedef bf16_t v16bf __attribute__((ext_vector_type(16)));
typedef float  v8f   __attribute__((ext_vector_type(8)));

__device__ __forceinline__ float lrelu(float x) { return x > 0.f ? x : 0.2f * x; }

__device__ __forceinline__ void edge_sd(const int* __restrict__ ei, int e, int& s, int& d) {
  if (e < EE) { s = ei[e]; d = ei[EE + e]; }
  else        { s = e - EE; d = s; }        // self-loops appended after real edges
}

// sign-aware float atomic max (monotonic in both branches; init to -1e30)
__device__ __forceinline__ void atomicMaxF(float* addr, float v) {
  if (v >= 0.f) atomicMax((int*)addr, __float_as_int(v));
  else          atomicMin((unsigned int*)addr, __float_as_uint(v));
}

// ---------------- init accumulators ----------------
__global__ void k_init(float* agg1, float* m1, float* den1,
                       float* m2, float* den2, float* agg2) {
  int i = blockIdx.x * blockDim.x + threadIdx.x;
  if (i < NN * HD) agg1[i] = 0.f;
  if (i < NN * HH) { m1[i] = -1e30f; den1[i] = 0.f; }
  if (i < NN)      { m2[i] = -1e30f; den2[i] = 0.f; }
  if (i < NN * CC) agg2[i] = 0.f;
}

// ---------------- swizzle A = x into 16-bit A-fragment layout (16x32 per tile) --------
// lane<16: v0..3 = K kb..kb+7, v4..7 = K kb+16..kb+23 ; lane>=16: +8 on both halves
__global__ void k_swzA(const float* __restrict__ x, bf16_t* __restrict__ A) {
  int i = blockIdx.x * blockDim.x + threadIdx.x;
  if (i >= 625 * 4 * 32) return;
  int lane = i & 31, kc = (i >> 5) & 3, mt = i >> 7;
  int row = mt * 16 + (lane & 15);
  int kb = kc * 32 + (lane >> 4) * 8;
  bf16_t* out = A + (size_t)i * 16;
  const float* xr = x + (size_t)row * INC;
#pragma unroll
  for (int j = 0; j < 8; ++j) {
    out[j]     = (bf16_t)xr[kb + j];
    out[8 + j] = (bf16_t)xr[kb + 16 + j];
  }
}

// ---------------- swizzle B = W1^T into 16-bit B-fragment layout (32x16 per tile) -----
// lane<16: v0..3 = K kb..kb+7, v4..7 = K kb+16..kb+23 ; lane>=16: +8 on both halves; col = lane&15
__global__ void k_swzB(const float* __restrict__ W1, bf16_t* __restrict__ B) {
  int i = blockIdx.x * blockDim.x + threadIdx.x;
  if (i >= 32 * 4 * 32) return;
  int lane = i & 31, kc = (i >> 5) & 3, ct = i >> 7;
  int col = ct * 16 + (lane & 15);
  int kb = kc * 32 + (lane >> 4) * 8;
  bf16_t* out = B + (size_t)i * 16;
  const float* wr = W1 + (size_t)col * INC;   // B[k,col] = W1[col,k]
#pragma unroll
  for (int e = 0; e < 16; ++e) out[e] = (bf16_t)wr[kb + (e & 7) + (e >> 3) * 16];
}

// ---------------- GEMM1: h1 = x @ W1^T, one 16x16 tile per wave, WMMA bf16 ------------
__global__ void k_gemm1(const bf16_t* __restrict__ A, const bf16_t* __restrict__ B,
                        float* __restrict__ h1) {
  int wave = (blockIdx.x * blockDim.x + threadIdx.x) >> 5;  // 20000 waves exactly
  int lane = threadIdx.x & 31;
  int mt = wave >> 5;   // 625 row tiles
  int ct = wave & 31;   // 32 col tiles
  v8f acc = {0.f, 0.f, 0.f, 0.f, 0.f, 0.f, 0.f, 0.f};
#pragma unroll
  for (int kc = 0; kc < 4; ++kc) {
    v16bf a = *(const v16bf*)(A + (((size_t)mt * 4 + kc) * 32 + lane) * 16);
    v16bf b = *(const v16bf*)(B + (((size_t)ct * 4 + kc) * 32 + lane) * 16);
    acc = __builtin_amdgcn_wmma_f32_16x16x32_bf16(false, a, false, b,
                                                  (short)0, acc, false, false);
  }
  // C/D layout: lane<16 -> rows 0..7 col=lane ; lane>=16 -> rows 8..15 col=lane-16
  int colg = lane & 15;
  int rowb = (lane >> 4) * 8;
  float* outp = h1 + ((size_t)(mt * 16 + rowb)) * HD + ct * 16 + colg;
#pragma unroll
  for (int r = 0; r < 8; ++r) outp[(size_t)r * HD] = acc[r];
}

// ---------------- attention logits layer 1 ----------------
__global__ void k_alpha1(const float* __restrict__ h1, const float* __restrict__ asrc,
                         const float* __restrict__ adst,
                         float* __restrict__ as1, float* __restrict__ ad1) {
  int i = blockIdx.x * blockDim.x + threadIdx.x;
  if (i >= NN * HH) return;
  int n = i >> 3, h = i & 7;
  const float4* hp = (const float4*)(h1 + (size_t)n * HD + h * DD);
  const float4* sp = (const float4*)(asrc + h * DD);
  const float4* dp = (const float4*)(adst + h * DD);
  float s = 0.f, d = 0.f;
#pragma unroll
  for (int k = 0; k < 16; ++k) {
    float4 v = hp[k], a = sp[k], b = dp[k];
    s += v.x * a.x + v.y * a.y + v.z * a.z + v.w * a.w;
    d += v.x * b.x + v.y * b.y + v.z * b.z + v.w * b.w;
  }
  as1[i] = s; ad1[i] = d;
}

// ---------------- layer-1 edge passes ----------------
__global__ void k_emax1(const int* __restrict__ ei, const float* __restrict__ as1,
                        const float* __restrict__ ad1, float* __restrict__ m1) {
  int i = blockIdx.x * blockDim.x + threadIdx.x;
  if (i >= ENL * HH) return;
  int e = i >> 3, h = i & 7, s, d;
  edge_sd(ei, e, s, d);
  atomicMaxF(&m1[d * HH + h], lrelu(as1[s * HH + h] + ad1[d * HH + h]));
}

__global__ void k_eden1(const int* __restrict__ ei, const float* __restrict__ as1,
                        const float* __restrict__ ad1, const float* __restrict__ m1,
                        float* __restrict__ den1, float* __restrict__ wbuf) {
  int i = blockIdx.x * blockDim.x + threadIdx.x;
  if (i >= ENL * HH) return;
  int e = i >> 3, h = i & 7, s, d;
  edge_sd(ei, e, s, d);
  float v = lrelu(as1[s * HH + h] + ad1[d * HH + h]);
  float ex = __expf(v - m1[d * HH + h]);
  wbuf[i] = ex;
  atomicAdd(&den1[d * HH + h], ex);
}

__global__ void k_eagg1(const int* __restrict__ ei, const float* __restrict__ h1,
                        const float* __restrict__ wbuf, const float* __restrict__ den1,
                        float* __restrict__ agg1) {
  long long i = (long long)blockIdx.x * blockDim.x + threadIdx.x;
  if (i >= (long long)ENL * HH * 16) return;
  int q = (int)(i & 15), h = (int)((i >> 4) & 7), e = (int)(i >> 7), s, d;
  edge_sd(ei, e, s, d);
  float w = wbuf[e * HH + h] / (den1[d * HH + h] + 1e-16f);
  float4 m = ((const float4*)(h1 + (size_t)s * HD + h * DD))[q];
  float* ap = agg1 + (size_t)d * HD + h * DD + q * 4;
  atomicAdd(ap + 0, w * m.x);
  atomicAdd(ap + 1, w * m.y);
  atomicAdd(ap + 2, w * m.z);
  atomicAdd(ap + 3, w * m.w);
}

// ---------------- bias + ELU (in place: agg1 -> h2) ----------------
__global__ void k_elu(float* __restrict__ agg1, const float* __restrict__ b1) {
  int i = blockIdx.x * blockDim.x + threadIdx.x;
  if (i >= NN * HD) return;
  float v = agg1[i] + b1[i & (HD - 1)];
  agg1[i] = v > 0.f ? v : (__expf(v) - 1.f);
}

// ---------------- layer-2 GEMV + logits (one wave per node, wave32 shuffles) ---------
__global__ void k_gemv2(const float* __restrict__ h2, const float* __restrict__ W2,
                        const float* __restrict__ a_s, const float* __restrict__ a_d,
                        float* __restrict__ g2, float* __restrict__ as2,
                        float* __restrict__ ad2) {
  int wave = (blockIdx.x * blockDim.x + threadIdx.x) >> 5;
  int lane = threadIdx.x & 31;
  if (wave >= NN) return;
  const float4* hp = (const float4*)(h2 + (size_t)wave * HD);
  const float4* w0 = (const float4*)(W2);
  const float4* w1 = (const float4*)(W2 + HD);
  float s0 = 0.f, s1 = 0.f;
#pragma unroll
  for (int k = 0; k < 4; ++k) {
    int j = k * 32 + lane;
    float4 v = hp[j], a = w0[j], b = w1[j];
    s0 += v.x * a.x + v.y * a.y + v.z * a.z + v.w * a.w;
    s1 += v.x * b.x + v.y * b.y + v.z * b.z + v.w * b.w;
  }
  for (int off = 16; off > 0; off >>= 1) {
    s0 += __shfl_down(s0, off, 32);
    s1 += __shfl_down(s1, off, 32);
  }
  if (lane == 0) {
    g2[wave * 2] = s0; g2[wave * 2 + 1] = s1;
    as2[wave] = s0 * a_s[0] + s1 * a_s[1];
    ad2[wave] = s0 * a_d[0] + s1 * a_d[1];
  }
}

// ---------------- layer-2 edge passes (1 head, C=2) ----------------
__global__ void k_emax2(const int* __restrict__ ei, const float* __restrict__ as2,
                        const float* __restrict__ ad2, float* __restrict__ m2) {
  int e = blockIdx.x * blockDim.x + threadIdx.x;
  if (e >= ENL) return;
  int s, d; edge_sd(ei, e, s, d);
  atomicMaxF(&m2[d], lrelu(as2[s] + ad2[d]));
}

__global__ void k_eden2(const int* __restrict__ ei, const float* __restrict__ as2,
                        const float* __restrict__ ad2, const float* __restrict__ m2,
                        float* __restrict__ den2, float* __restrict__ wbuf) {
  int e = blockIdx.x * blockDim.x + threadIdx.x;
  if (e >= ENL) return;
  int s, d; edge_sd(ei, e, s, d);
  float ex = __expf(lrelu(as2[s] + ad2[d]) - m2[d]);
  wbuf[e] = ex;
  atomicAdd(&den2[d], ex);
}

__global__ void k_eagg2(const int* __restrict__ ei, const float* __restrict__ g2,
                        const float* __restrict__ wbuf, const float* __restrict__ den2,
                        float* __restrict__ agg2) {
  int e = blockIdx.x * blockDim.x + threadIdx.x;
  if (e >= ENL) return;
  int s, d; edge_sd(ei, e, s, d);
  float w = wbuf[e] / (den2[d] + 1e-16f);
  atomicAdd(&agg2[d * 2 + 0], w * g2[s * 2 + 0]);
  atomicAdd(&agg2[d * 2 + 1], w * g2[s * 2 + 1]);
}

__global__ void k_final(const float* __restrict__ agg2, const float* __restrict__ b2,
                        float* __restrict__ out) {
  int i = blockIdx.x * blockDim.x + threadIdx.x;
  if (i >= NN * CC) return;
  out[i] = agg2[i] + b2[i & 1];
}

// ============================================================================
extern "C" void kernel_launch(void* const* d_in, const int* in_sizes, int n_in,
                              void* d_out, int out_size, void* d_ws, size_t ws_size,
                              hipStream_t stream) {
  const float* x     = (const float*)d_in[0];
  const int*   ei    = (const int*)  d_in[1];
  const float* W1    = (const float*)d_in[2];
  const float* asrc1 = (const float*)d_in[3];
  const float* adst1 = (const float*)d_in[4];
  const float* b1    = (const float*)d_in[5];
  const float* W2    = (const float*)d_in[6];
  const float* asrc2 = (const float*)d_in[7];
  const float* adst2 = (const float*)d_in[8];
  const float* b2    = (const float*)d_in[9];
  float* out = (float*)d_out;

  char* ws = (char*)d_ws;
  size_t off = 0;
  auto carve = [&](size_t bytes) -> void* {
    void* p = ws + off;
    off = (off + bytes + 255) & ~(size_t)255;
    return p;
  };
  bf16_t* Abf  = (bf16_t*)carve((size_t)625 * 4 * 32 * 16 * sizeof(bf16_t));
  bf16_t* Bbf  = (bf16_t*)carve((size_t)32 * 4 * 32 * 16 * sizeof(bf16_t));
  float* h1    = (float*)carve((size_t)NN * HD * 4);
  float* agg1  = (float*)carve((size_t)NN * HD * 4);
  float* as1   = (float*)carve((size_t)NN * HH * 4);
  float* ad1   = (float*)carve((size_t)NN * HH * 4);
  float* m1    = (float*)carve((size_t)NN * HH * 4);
  float* den1  = (float*)carve((size_t)NN * HH * 4);
  float* wbuf  = (float*)carve((size_t)ENL * HH * 4);
  float* g2    = (float*)carve((size_t)NN * 2 * 4);
  float* as2   = (float*)carve((size_t)NN * 4);
  float* ad2   = (float*)carve((size_t)NN * 4);
  float* m2    = (float*)carve((size_t)NN * 4);
  float* den2  = (float*)carve((size_t)NN * 4);
  float* agg2  = (float*)carve((size_t)NN * 2 * 4);
  (void)in_sizes; (void)n_in; (void)out_size; (void)ws_size;

  const int T = 256;
  auto g = [](long long n, int t) { return (unsigned)((n + t - 1) / t); };

  k_init <<<g((long long)NN * HD, T), T, 0, stream>>>(agg1, m1, den1, m2, den2, agg2);
  k_swzA <<<g(625 * 4 * 32, T),       T, 0, stream>>>(x, Abf);
  k_swzB <<<g(32 * 4 * 32, T),        T, 0, stream>>>(W1, Bbf);
  k_gemm1<<<2500,                     T, 0, stream>>>(Abf, Bbf, h1);   // 20000 waves
  k_alpha1<<<g((long long)NN * HH, T), T, 0, stream>>>(h1, asrc1, adst1, as1, ad1);
  k_emax1<<<g((long long)ENL * HH, T), T, 0, stream>>>(ei, as1, ad1, m1);
  k_eden1<<<g((long long)ENL * HH, T), T, 0, stream>>>(ei, as1, ad1, m1, den1, wbuf);
  k_eagg1<<<g((long long)ENL * HH * 16, T), T, 0, stream>>>(ei, h1, wbuf, den1, agg1);
  k_elu  <<<g((long long)NN * HD, T), T, 0, stream>>>(agg1, b1);
  k_gemv2<<<1250,                     T, 0, stream>>>(agg1, W2, asrc2, adst2, g2, as2, ad2);
  k_emax2<<<g(ENL, T),                T, 0, stream>>>(ei, as2, ad2, m2);
  k_eden2<<<g(ENL, T),                T, 0, stream>>>(ei, as2, ad2, m2, den2, wbuf);
  k_eagg2<<<g(ENL, T),                T, 0, stream>>>(ei, g2, wbuf, den2, agg2);
  k_final<<<g((long long)NN * CC, T), T, 0, stream>>>(agg2, b2, out);
}
